// TimesBlock_81252191306604
// MI455X (gfx1250) — matrix operand order repaired and checked by reading, not compiled, not run on hardware
//
#include <hip/hip_runtime.h>
#include <hip/hip_bf16.h>
#include <math.h>

typedef __attribute__((ext_vector_type(16))) __bf16 v16bf;
typedef __attribute__((ext_vector_type(8)))  __bf16 v8bf;
typedef __attribute__((ext_vector_type(8)))  float  v8f;

#define B_   64
#define L_   1024
#define D_   256
#define C_   128
#define HMAX 1536   // max T*p = 1533 (p=511), rounded to 16

// ---------------------------------------------------------------------------
// 1) channel mean + f32->bf16 convert.  One wave per (b,l) row of 256 floats.
// ---------------------------------------------------------------------------
__global__ __launch_bounds__(256)
void prep_kernel(const float* __restrict__ x, __bf16* __restrict__ xbf,
                 float* __restrict__ m)
{
    const int lane = threadIdx.x & 31;
    const int row  = blockIdx.x * (blockDim.x >> 5) + (threadIdx.x >> 5);
    if (row >= B_ * L_) return;
    const float* rp = x   + (size_t)row * D_;
    __bf16*      bp = xbf + (size_t)row * D_;
    float s = 0.f;
    #pragma unroll
    for (int i = 0; i < 8; ++i) {
        float v = rp[lane + 32 * i];
        s += v;
        bp[lane + 32 * i] = (__bf16)v;
    }
    #pragma unroll
    for (int off = 16; off > 0; off >>= 1) s += __shfl_xor(s, off, 32);
    if (lane == 0) m[row] = s * (1.0f / 256.0f);
}

// ---------------------------------------------------------------------------
// 2) pre-pack W1/W2 (f32, OIHW) into bf16 A-fragment lane order; zero zpad.
//    Layout: p[(((tap*KC+kc)*MC + mc)*32 + lane)*16 + i]
//    K-map (16-bit A 16x32): h=lane>>4:
//      K = 16h + i  (contiguous, matches B loads)
// ---------------------------------------------------------------------------
__global__ __launch_bounds__(256)
void pack_kernel(const float* __restrict__ W1, const float* __restrict__ W2,
                 __bf16* __restrict__ pW1, __bf16* __restrict__ pW2,
                 __bf16* __restrict__ zpad)
{
    const int idx = blockIdx.x * blockDim.x + threadIdx.x;
    if (idx < 256) zpad[idx] = (__bf16)0.0f;     // 512-byte zero row
    const int N1  = 9 * 8 * 8 * 512;             // W1 packed elems
    if (idx >= 2 * N1) return;
    const bool isW2 = (idx >= N1);
    const int  t    = isW2 ? idx - N1 : idx;
    const int  MC   = isW2 ? 16 : 8;             // M chunks (out-ch / 16)
    const int  KC   = isW2 ? 4  : 8;             // K chunks (in-ch / 32)

    const int i    = t & 15;
    const int lane = (t >> 4) & 31;
    const int rest = t >> 9;
    const int mc   = rest % MC;
    const int tk   = rest / MC;
    const int kc   = tk % KC;
    const int tap  = tk / KC;

    const int mrow = lane & 15, h = lane >> 4;
    const int j = i >> 1, hi = i & 1;
    const int K = 16 * h + i;
    const int kidx = kc * 32 + K;                // input channel
    const int mo   = mc * 16 + mrow;             // output channel

    float w;
    if (isW2) w = W2[((size_t)mo * C_ + kidx) * 9 + tap];
    else      w = W1[((size_t)mo * D_ + kidx) * 9 + tap];
    (isW2 ? pW2 : pW1)[t] = (__bf16)w;
}

// ---------------------------------------------------------------------------
// 3) per-batch DFT magnitude of the mean series + top-3 (period = idx+1).
//    Phase reduced mod 1024 so __sincosf args stay in [-2pi, 0].
// ---------------------------------------------------------------------------
__global__ __launch_bounds__(256)
void dft_topk_kernel(const float* __restrict__ m, int* __restrict__ periods)
{
    __shared__ float sm[L_];
    __shared__ float mag[513];
    const int b = blockIdx.x, tid = threadIdx.x;
    for (int i = tid; i < L_; i += 256) sm[i] = m[(size_t)b * L_ + i];
    __syncthreads();
    for (int k = tid; k <= 512; k += 256) {
        if (k == 0) continue;
        float re = 0.f, im = 0.f;
        for (int l = 0; l < L_; ++l) {
            const int ph = (k * l) & 1023;
            float s, c;
            __sincosf(-6.283185307179586f * (float)ph * (1.0f / 1024.0f), &s, &c);
            re += sm[l] * c;
            im += sm[l] * s;
        }
        mag[k] = re * re + im * im;     // ordering == |fft|
    }
    __syncthreads();
    if (tid == 0) {
        int i0 = 1, i1 = 1, i2 = 1;
        float v0 = -1.f, v1 = -1.f, v2 = -1.f;
        for (int k = 1; k <= 512; ++k) {   // strict '>' == top_k first-occurrence
            const float v = mag[k];
            if (v > v0)      { v2=v1; i2=i1; v1=v0; i1=i0; v0=v; i0=k; }
            else if (v > v1) { v2=v1; i2=i1; v1=v;  i1=k; }
            else if (v > v2) { v2=v;  i2=k; }
        }
        periods[b*3+0] = i0 + 1; periods[b*3+1] = i1 + 1; periods[b*3+2] = i2 + 1;
    }
}

// ---------------------------------------------------------------------------
// 4) conv1 (D->C, 3x3, implicit GEMM by taps) + bias + exact GELU -> h (bf16)
//    One wave: 2M x 2N = 32(out-ch) x 32(spatial), 4 independent acc chains.
//    Invalid halo lanes read from a zeroed row (no exec-mask branching).
// ---------------------------------------------------------------------------
__global__ __launch_bounds__(128)
void conv1_kernel(const __bf16* __restrict__ xbf, const __bf16* __restrict__ pW1,
                  const float* __restrict__ b1, const int* __restrict__ periods,
                  const __bf16* __restrict__ zrow, __bf16* __restrict__ hbuf)
{
    const int lane = threadIdx.x & 31;
    const int wave = threadIdx.x >> 5;
    const int bs   = blockIdx.z;                 // b*3 + slot
    const int b    = bs / 3;
    const int slot = bs - 3 * b;
    const int c0   = wave * 2;                   // out-channel chunks {c0, c0+1}
    const int p    = periods[b * 3 + slot];
    if (p < 2) return;
    const int T  = (L_ + p - 1) / p;
    const int Tp = T * p;
    const int n0 = blockIdx.x * 32;
    if (n0 >= Tp) return;

    const int hf = lane >> 4;
    const int na = n0 + (lane & 15);             // spatial columns of the 2 N-tiles
    const int nb = na + 16;
    const unsigned ta_ = (unsigned)na / (unsigned)p;
    const unsigned tb_ = (unsigned)nb / (unsigned)p;
    const int qa = na - (int)ta_ * p;
    const int qb = nb - (int)tb_ * p;

    v8f acc00 = {}, acc01 = {}, acc10 = {}, acc11 = {};   // [M-chunk][N-tile]
    const __bf16* xb = xbf + (size_t)b * (L_ * D_);
    #pragma unroll
    for (int ky = 0; ky < 3; ++ky) {
        #pragma unroll
        for (int kx = 0; kx < 3; ++kx) {
            const int tap = ky * 3 + kx;
            const int tta = (int)ta_ + ky - 1, qqa = qa + kx - 1;
            const int ttb = (int)tb_ + ky - 1, qqb = qb + kx - 1;
            const int lpa = tta * p + qqa;
            const int lpb = ttb * p + qqb;
            const bool va = (tta >= 0) & (tta < T) & (qqa >= 0) & (qqa < p) & (lpa < L_);
            const bool vb = (ttb >= 0) & (ttb < T) & (qqb >= 0) & (qqb < p) & (lpb < L_);
            const __bf16* browa = (va ? xb + (size_t)lpa * D_ : zrow) + hf * 16;
            const __bf16* browb = (vb ? xb + (size_t)lpb * D_ : zrow) + hf * 16;
            #pragma unroll
            for (int kc = 0; kc < 8; ++kc) {
                v16bf bba = *(const v16bf*)(browa + kc * 32);
                v16bf bbb = *(const v16bf*)(browb + kc * 32);
                v16bf a0  = *(const v16bf*)(pW1 + (((size_t)(tap*8+kc)*8 + c0    )*512 + lane*16));
                v16bf a1  = *(const v16bf*)(pW1 + (((size_t)(tap*8+kc)*8 + c0 + 1)*512 + lane*16));
                acc00 = __builtin_amdgcn_wmma_f32_16x16x32_bf16(false, a0, false, bba, (short)0, acc00, false, false);
                acc01 = __builtin_amdgcn_wmma_f32_16x16x32_bf16(false, a0, false, bbb, (short)0, acc01, false, false);
                acc10 = __builtin_amdgcn_wmma_f32_16x16x32_bf16(false, a1, false, bba, (short)0, acc10, false, false);
                acc11 = __builtin_amdgcn_wmma_f32_16x16x32_bf16(false, a1, false, bbb, (short)0, acc11, false, false);
            }
        }
    }
    // D layout: lane = column n, rows c = cch*16 + 8*hf + r  (contiguous in [n][c])
    #pragma unroll
    for (int w = 0; w < 2; ++w) {                 // N-tile
        const int n = w ? nb : na;
        __bf16* hrow = hbuf + ((size_t)bs * HMAX + n) * C_;
        #pragma unroll
        for (int u = 0; u < 2; ++u) {             // M-chunk
            const v8f& a = u ? (w ? acc11 : acc10) : (w ? acc01 : acc00);
            const int cbase = (c0 + u) * 16 + hf * 8;
            v8bf st;
            #pragma unroll
            for (int r = 0; r < 8; ++r) {
                float v = a[r] + b1[cbase + r];
                v = 0.5f * v * (1.0f + erff(v * 0.70710678118654752f));   // exact GELU
                st[r] = (__bf16)v;
            }
            *(v8bf*)(hrow + cbase) = st;
        }
    }
}

// ---------------------------------------------------------------------------
// 5) conv2 (C->D, 3x3) summed over 3 slots, /3, +bias, +residual -> out (f32)
//    One wave: 2M x 2N = 32(out-ch) x 32(l), 4 independent acc chains.
// ---------------------------------------------------------------------------
__global__ __launch_bounds__(128)
void conv2_kernel(const float* __restrict__ x, const __bf16* __restrict__ hbuf,
                  const __bf16* __restrict__ pW2, const float* __restrict__ b2,
                  const int* __restrict__ periods, const __bf16* __restrict__ zrow,
                  float* __restrict__ out)
{
    const int lane = threadIdx.x & 31;
    const int wave = threadIdx.x >> 5;
    const int b    = blockIdx.z;
    const int d0   = blockIdx.y * 8 + wave * 2;  // out-channel chunks {d0, d0+1} of 16
    const int hf   = lane >> 4;
    const int la   = blockIdx.x * 32 + (lane & 15);
    const int lb   = la + 16;

    v8f acc00 = {}, acc01 = {}, acc10 = {}, acc11 = {};   // [M-chunk][N-tile]
    for (int slot = 0; slot < 3; ++slot) {
        const int p = periods[b * 3 + slot];
        if (p < 2) continue;
        const int T = (L_ + p - 1) / p;
        const unsigned ta_ = (unsigned)la / (unsigned)p;
        const unsigned tb_ = (unsigned)lb / (unsigned)p;
        const int qa = la - (int)ta_ * p;
        const int qb = lb - (int)tb_ * p;
        const __bf16* hb = hbuf + (size_t)(b * 3 + slot) * HMAX * C_;
        #pragma unroll
        for (int ky = 0; ky < 3; ++ky) {
            #pragma unroll
            for (int kx = 0; kx < 3; ++kx) {
                const int tap = ky * 3 + kx;
                const int tta = (int)ta_ + ky - 1, qqa = qa + kx - 1;
                const int ttb = (int)tb_ + ky - 1, qqb = qb + kx - 1;
                const bool va = (tta >= 0) & (tta < T) & (qqa >= 0) & (qqa < p);
                const bool vb = (ttb >= 0) & (ttb < T) & (qqb >= 0) & (qqb < p);
                const int npa = tta * p + qqa;
                const int npb = ttb * p + qqb;
                const __bf16* browa = (va ? hb + (size_t)npa * C_ : zrow) + hf * 16;
                const __bf16* browb = (vb ? hb + (size_t)npb * C_ : zrow) + hf * 16;
                #pragma unroll
                for (int kc = 0; kc < 4; ++kc) {
                    v16bf bba = *(const v16bf*)(browa + kc * 32);
                    v16bf bbb = *(const v16bf*)(browb + kc * 32);
                    v16bf a0  = *(const v16bf*)(pW2 + (((size_t)(tap*4+kc)*16 + d0    )*512 + lane*16));
                    v16bf a1  = *(const v16bf*)(pW2 + (((size_t)(tap*4+kc)*16 + d0 + 1)*512 + lane*16));
                    acc00 = __builtin_amdgcn_wmma_f32_16x16x32_bf16(false, a0, false, bba, (short)0, acc00, false, false);
                    acc01 = __builtin_amdgcn_wmma_f32_16x16x32_bf16(false, a0, false, bbb, (short)0, acc01, false, false);
                    acc10 = __builtin_amdgcn_wmma_f32_16x16x32_bf16(false, a1, false, bba, (short)0, acc10, false, false);
                    acc11 = __builtin_amdgcn_wmma_f32_16x16x32_bf16(false, a1, false, bbb, (short)0, acc11, false, false);
                }
            }
        }
    }
    #pragma unroll
    for (int w = 0; w < 2; ++w) {                 // N-tile (l column)
        const size_t row = (size_t)b * L_ + (w ? lb : la);
        #pragma unroll
        for (int u = 0; u < 2; ++u) {             // M-chunk
            const v8f& a = u ? (w ? acc11 : acc10) : (w ? acc01 : acc00);
            const int dbase = (d0 + u) * 16 + hf * 8;
            const v8f xin = *(const v8f*)(x + row * D_ + dbase);
            v8f res;
            #pragma unroll
            for (int r = 0; r < 8; ++r)
                res[r] = xin[r] + a[r] * (1.0f / 3.0f) + b2[dbase + r];
            *(v8f*)(out + row * D_ + dbase) = res;
        }
    }
}

// ---------------------------------------------------------------------------
// 6) LayerNorm over D=256 in place on d_out.  One wave per row.
// ---------------------------------------------------------------------------
__global__ __launch_bounds__(256)
void ln_kernel(float* __restrict__ out, const float* __restrict__ gamma,
               const float* __restrict__ beta)
{
    const int lane = threadIdx.x & 31;
    const int row  = blockIdx.x * (blockDim.x >> 5) + (threadIdx.x >> 5);
    if (row >= B_ * L_) return;
    float* rp = out + (size_t)row * D_;
    float v[8], s = 0.f, s2 = 0.f;
    #pragma unroll
    for (int i = 0; i < 8; ++i) {
        v[i] = rp[lane + 32 * i];
        s += v[i]; s2 += v[i] * v[i];
    }
    #pragma unroll
    for (int off = 16; off > 0; off >>= 1) {
        s  += __shfl_xor(s,  off, 32);
        s2 += __shfl_xor(s2, off, 32);
    }
    const float mu   = s * (1.0f / 256.0f);
    const float var  = s2 * (1.0f / 256.0f) - mu * mu;
    const float rstd = rsqrtf(var + 1e-5f);
    #pragma unroll
    for (int i = 0; i < 8; ++i) {
        const int d = lane + 32 * i;
        rp[d] = (v[i] - mu) * rstd * gamma[d] + beta[d];
    }
}

// ---------------------------------------------------------------------------
extern "C" void kernel_launch(void* const* d_in, const int* in_sizes, int n_in,
                              void* d_out, int out_size, void* d_ws, size_t ws_size,
                              hipStream_t stream)
{
    (void)in_sizes; (void)n_in; (void)out_size; (void)ws_size;
    const float* x     = (const float*)d_in[0];
    const float* W1    = (const float*)d_in[1];
    const float* b1    = (const float*)d_in[2];
    const float* W2    = (const float*)d_in[3];
    const float* b2    = (const float*)d_in[4];
    const float* gamma = (const float*)d_in[5];
    const float* beta  = (const float*)d_in[6];
    float* out = (float*)d_out;

    char* ws = (char*)d_ws;
    size_t off = 0;
    auto alloc = [&](size_t bytes) -> void* {
        off = (off + 255) & ~(size_t)255;
        void* p = ws + off; off += bytes; return p;
    };
    __bf16* xbf     = (__bf16*)alloc((size_t)B_ * L_ * D_ * 2);        // 33.6 MB
    float*  m       = (float*) alloc((size_t)B_ * L_ * 4);             // 0.26 MB
    int*    periods = (int*)   alloc((size_t)B_ * 3 * 4);
    __bf16* pW1     = (__bf16*)alloc((size_t)9 * 8 * 8  * 512 * 2);    // 0.59 MB
    __bf16* pW2     = (__bf16*)alloc((size_t)9 * 4 * 16 * 512 * 2);    // 0.59 MB
    __bf16* zpad    = (__bf16*)alloc((size_t)256 * 2);                 // 512 B zeros
    __bf16* hbuf    = (__bf16*)alloc((size_t)B_ * 3 * HMAX * C_ * 2);  // 75.5 MB

    prep_kernel    <<<dim3((B_ * L_) / 8), dim3(256), 0, stream>>>(x, xbf, m);
    pack_kernel    <<<dim3((2 * 9 * 8 * 8 * 512 + 255) / 256), dim3(256), 0, stream>>>(W1, W2, pW1, pW2, zpad);
    dft_topk_kernel<<<dim3(B_), dim3(256), 0, stream>>>(m, periods);
    conv1_kernel   <<<dim3(HMAX / 32, 1, B_ * 3), dim3(128), 0, stream>>>(xbf, pW1, b1, periods, zpad, hbuf);
    conv2_kernel   <<<dim3(L_ / 32, 2, B_), dim3(128), 0, stream>>>(x, hbuf, pW2, b2, periods, zpad, out);
    ln_kernel      <<<dim3((B_ * L_) / 8), dim3(256), 0, stream>>>(out, gamma, beta);
}
